// CrossMultiHeadAttention_66709432042331
// MI455X (gfx1250) — matrix operand order repaired and checked by reading, compiled, not run on hardware
//
#include <hip/hip_runtime.h>
#include <hip/hip_bf16.h>

// Problem constants (match reference)
#define BB   128
#define SS   256
#define DD   512
#define HH   8
#define DH   64
#define MM   (BB * SS)   // 32768 rows
#define DCAT 192         // 3 * DH concatenated head dim for fused scores

typedef __bf16 bf16;
typedef __attribute__((ext_vector_type(16))) __bf16 bf16x16;
typedef __attribute__((ext_vector_type(8)))  float  f32x8;

union Frag32B { uint4 u[2]; bf16x16 v; };

__device__ __forceinline__ f32x8 zero8() {
  f32x8 z;
#pragma unroll
  for (int i = 0; i < 8; ++i) z[i] = 0.0f;
  return z;
}

__device__ __forceinline__ f32x8 wmma_bf16(bf16x16 a, bf16x16 b, f32x8 c) {
  // 8 args: (neg_a, A, neg_b, B, c_mod, C, reuse_a, reuse_b)
  return __builtin_amdgcn_wmma_f32_16x16x32_bf16(false, a, false, b, (short)0, c,
                                                 false, false);
}

// A-operand 16x32 bf16: lane = m + 16*kh holds K = kb..kb+7 and kb+16..kb+23,
// kb = k0 + 8*kh  (ISA 7.12.2 "16-bit A-Matrix 16x32")
__device__ __forceinline__ bf16x16 load_a(const bf16* rowptr, int k0, int lane) {
  int kb = k0 + ((lane >> 4) << 3);
  Frag32B f;
  f.u[0] = *reinterpret_cast<const uint4*>(rowptr + kb);
  f.u[1] = *reinterpret_cast<const uint4*>(rowptr + kb + 16);
  return f.v;
}

// B-operand 32x16 bf16: lane = n + 16*kh holds K = kb..kb+7 and kb+16..kb+23,
// kb = k0 + 8*kh (same K order as the A operand)
__device__ __forceinline__ bf16x16 load_b(const bf16* rowptr, int k0, int lane) {
  int kb = k0 + ((lane >> 4) << 3);
  Frag32B f;
  f.u[0] = *reinterpret_cast<const uint4*>(rowptr + kb);
  f.u[1] = *reinterpret_cast<const uint4*>(rowptr + kb + 16);
  return f.v;
}

__device__ __forceinline__ float hmax16(float x) {
  x = fmaxf(x, __shfl_xor(x, 1));
  x = fmaxf(x, __shfl_xor(x, 2));
  x = fmaxf(x, __shfl_xor(x, 4));
  x = fmaxf(x, __shfl_xor(x, 8));
  return x;
}
__device__ __forceinline__ float hsum16(float x) {
  x += __shfl_xor(x, 1);
  x += __shfl_xor(x, 2);
  x += __shfl_xor(x, 4);
  x += __shfl_xor(x, 8);
  return x;
}

// ---------------------------------------------------------------- pack kernels
__global__ void pack_inputs_kernel(const float* __restrict__ xin,
                                   const float* __restrict__ xattr,
                                   const float* __restrict__ xpos,
                                   bf16* __restrict__ oin,
                                   bf16* __restrict__ oattr,
                                   bf16* __restrict__ opos, int n) {
  int i = blockIdx.x * blockDim.x + threadIdx.x;
  if (i < n) {
    oin[i]   = (bf16)xin[i];
    oattr[i] = (bf16)xattr[i];
    opos[i]  = (bf16)xpos[i];
  }
}

// Transpose all 8 DxD f32 weights into N-major bf16 (Wt[n*D + k] = W[k*D + n])
__global__ void pack_weights_kernel(const float* __restrict__ w0, const float* __restrict__ w1,
                                    const float* __restrict__ w2, const float* __restrict__ w3,
                                    const float* __restrict__ w4, const float* __restrict__ w5,
                                    const float* __restrict__ w6, const float* __restrict__ w7,
                                    bf16* __restrict__ wt) {
  int i = blockIdx.x * blockDim.x + threadIdx.x;
  if (i >= DD * DD) return;
  int nn = i >> 9;       // output row (N)
  int kk = i & (DD - 1); // output col (K)
  const float* srcs[8] = {w0, w1, w2, w3, w4, w5, w6, w7};
#pragma unroll
  for (int j = 0; j < 8; ++j)
    wt[(size_t)j * DD * DD + (size_t)nn * DD + kk] = (bf16)srcs[j][(size_t)kk * DD + nn];
}

// ----------------------------------------------------- projection GEMM (WMMA)
// C(M x 512) = A(M x 512) @ Wt^T + bias ; one wave computes a 16x64 tile.
// mode 0: bf16 row-major out[m*512+n]
// mode 1: Kcat layout  [B,H,S,192] at column offset coloff
// mode 2: Vt layout    [B,H,64,S]  (transposed V for ctx GEMM B-operand)
__global__ void __launch_bounds__(32) proj_gemm_kernel(
    const bf16* __restrict__ A, const bf16* __restrict__ Wt,
    const float* __restrict__ bias, bf16* __restrict__ out,
    int mode, int coloff) {
  int m0 = blockIdx.x * 16;
  int n0 = blockIdx.y * 64;
  int lane = threadIdx.x;
  const bf16* arow = A + (size_t)(m0 + (lane & 15)) * DD;
  f32x8 acc[4];
#pragma unroll
  for (int t = 0; t < 4; ++t) acc[t] = zero8();

  for (int k0 = 0; k0 < DD; k0 += 32) {
    bf16x16 a = load_a(arow, k0, lane);
#pragma unroll
    for (int t = 0; t < 4; ++t) {
      const bf16* brow = Wt + (size_t)(n0 + t * 16 + (lane & 15)) * DD;
      acc[t] = wmma_bf16(a, load_b(brow, k0, lane), acc[t]);
    }
  }

  int col = lane & 15;
  int rbase = (lane >> 4) << 3;  // C-layout: row = vgpr + 8*(lane>=16)
#pragma unroll
  for (int t = 0; t < 4; ++t) {
    int n = n0 + t * 16 + col;
    float bv = bias[n];
#pragma unroll
    for (int v = 0; v < 8; ++v) {
      int m = m0 + rbase + v;
      float val = acc[t][v] + bv;
      int bb = m >> 8, ss = m & (SS - 1);
      int hh = n >> 6, cc = n & (DH - 1);
      if (mode == 0) {
        out[(size_t)m * DD + n] = (bf16)val;
      } else if (mode == 1) {
        out[((size_t)(bb * HH + hh) * SS + ss) * DCAT + coloff + cc] = (bf16)val;
      } else {
        out[((size_t)(bb * HH + hh) * DH + cc) * SS + ss] = (bf16)val;
      }
    }
  }
}

// --------------------------------------------------------------------- Q mix
// Qcat[...,0:64]=w00*Qi+w10*Qp+w20*Qa, [...,64:128]=col1, [...,128:192]=col2
__global__ void qmix_kernel(const bf16* __restrict__ Qi, const bf16* __restrict__ Qp,
                            const bf16* __restrict__ Qa, const float* __restrict__ w,
                            bf16* __restrict__ Qcat) {
  size_t i = (size_t)blockIdx.x * blockDim.x + threadIdx.x;
  if (i >= (size_t)MM * DD) return;
  int n = (int)(i & (DD - 1));
  int m = (int)(i >> 9);
  int hh = n >> 6, cc = n & (DH - 1);
  int bb = m >> 8, ss = m & (SS - 1);
  float qi = (float)Qi[i], qp = (float)Qp[i], qa = (float)Qa[i];
  size_t base = ((size_t)(bb * HH + hh) * SS + ss) * DCAT + cc;
  Qcat[base +   0] = (bf16)(w[0] * qi + w[3] * qp + w[6] * qa);
  Qcat[base +  64] = (bf16)(w[1] * qi + w[4] * qp + w[7] * qa);
  Qcat[base + 128] = (bf16)(w[2] * qi + w[5] * qp + w[8] * qa);
}

// ---------------------------------------------------------- fused attention
// Per wave: 16 q-rows. scores(16x256) = Qcat(16x192) @ Kcat^T, scale+mask,
// softmax, P->LDS bf16, ctx(16x64) = P @ V via Vt rows.
__global__ void __launch_bounds__(128) attention_kernel(
    const bf16* __restrict__ Qcat, const bf16* __restrict__ Kcat,
    const bf16* __restrict__ Vt, const float* __restrict__ mask,
    bf16* __restrict__ ctx) {
  int wg = blockIdx.x;
  int qblk = wg & 3;        // S/64 = 4 q-blocks
  int bh = wg >> 2;         // b*H + h
  int b = bh >> 3;
  int h = bh & 7;
  int wave = threadIdx.x >> 5;
  int lane = threadIdx.x & 31;
  int q0 = qblk * 64 + wave * 16;

  const bf16*  Qb = Qcat + (size_t)bh * SS * DCAT;
  const bf16*  Kb = Kcat + (size_t)bh * SS * DCAT;
  const bf16*  Vb = Vt   + (size_t)bh * DH * SS;
  const float* mb = mask + (size_t)b * SS * SS;

  __shared__ __align__(16) bf16 ldsP[4][16][SS];

  f32x8 s[16];
#pragma unroll
  for (int t = 0; t < 16; ++t) s[t] = zero8();

  const bf16* qrow = Qb + (size_t)(q0 + (lane & 15)) * DCAT;
  for (int k0 = 0; k0 < DCAT; k0 += 32) {
    bf16x16 a = load_a(qrow, k0, lane);
#pragma unroll
    for (int t = 0; t < 16; ++t) {
      const bf16* krow = Kb + (size_t)(t * 16 + (lane & 15)) * DCAT;
      s[t] = wmma_bf16(a, load_b(krow, k0, lane), s[t]);
    }
  }

  int col = lane & 15;
  int rbase = (lane >> 4) << 3;

  // scale (1/sqrt(64)) + attention mask
#pragma unroll
  for (int t = 0; t < 16; ++t) {
#pragma unroll
    for (int v = 0; v < 8; ++v) {
      int q = q0 + rbase + v;
      int k = t * 16 + col;
      s[t][v] = s[t][v] * 0.125f + mb[(size_t)q * SS + k];
    }
  }

  // row-wise softmax (row's 16 cols live in one 16-lane half; 16 tiles per lane)
#pragma unroll
  for (int v = 0; v < 8; ++v) {
    float mx = -3.0e38f;
#pragma unroll
    for (int t = 0; t < 16; ++t) mx = fmaxf(mx, s[t][v]);
    mx = hmax16(mx);
    float sum = 0.0f;
#pragma unroll
    for (int t = 0; t < 16; ++t) {
      float e = __expf(s[t][v] - mx);
      s[t][v] = e;
      sum += e;
    }
    sum = hsum16(sum);
    float r = 1.0f / sum;
    int row = rbase + v;
#pragma unroll
    for (int t = 0; t < 16; ++t)
      ldsP[wave][row][t * 16 + col] = (bf16)(s[t][v] * r);
  }
  __syncthreads();

  // ctx(16x64) = P(16x256) @ V(256x64); B-operand rows come from Vt
  f32x8 c[4];
#pragma unroll
  for (int t = 0; t < 4; ++t) c[t] = zero8();
  const bf16* prow = &ldsP[wave][lane & 15][0];
  for (int k0 = 0; k0 < SS; k0 += 32) {
    bf16x16 a = load_a(prow, k0, lane);
#pragma unroll
    for (int t = 0; t < 4; ++t) {
      const bf16* vrow = Vb + (size_t)(t * 16 + (lane & 15)) * SS;
      c[t] = wmma_bf16(a, load_b(vrow, k0, lane), c[t]);
    }
  }
#pragma unroll
  for (int t = 0; t < 4; ++t) {
#pragma unroll
    for (int v = 0; v < 8; ++v) {
      int q = q0 + rbase + v;
      int dcol = h * DH + t * 16 + col;
      ctx[((size_t)b * SS + q) * DD + dcol] = (bf16)c[t][v];
    }
  }
}

// --------------------------- output projection + residual + LayerNorm (fused)
// WG = 4 waves covers one 16-row x 512-col strip; wave owns 128 columns.
__global__ void __launch_bounds__(128) out_ln_kernel(
    const bf16* __restrict__ Ctx, const bf16* __restrict__ Wdt,
    const float* __restrict__ bias, const float* __restrict__ resid,
    const float* __restrict__ gamma, const float* __restrict__ beta,
    float* __restrict__ out) {
  int m0 = blockIdx.x * 16;
  int wave = threadIdx.x >> 5;
  int lane = threadIdx.x & 31;
  int n0 = wave * 128;

  __shared__ float s1[16][4];
  __shared__ float s2[16][4];

  const bf16* arow = Ctx + (size_t)(m0 + (lane & 15)) * DD;
  f32x8 acc[8];
#pragma unroll
  for (int t = 0; t < 8; ++t) acc[t] = zero8();

  for (int k0 = 0; k0 < DD; k0 += 32) {
    bf16x16 a = load_a(arow, k0, lane);
#pragma unroll
    for (int t = 0; t < 8; ++t) {
      const bf16* brow = Wdt + (size_t)(n0 + t * 16 + (lane & 15)) * DD;
      acc[t] = wmma_bf16(a, load_b(brow, k0, lane), acc[t]);
    }
  }

  int col = lane & 15;
  int rbase = (lane >> 4) << 3;

  // h = proj + bias + residual (kept in f32 accumulators)
#pragma unroll
  for (int t = 0; t < 8; ++t) {
    int n = n0 + t * 16 + col;
    float bv = bias[n];
#pragma unroll
    for (int v = 0; v < 8; ++v) {
      int m = m0 + rbase + v;
      acc[t][v] += bv + resid[(size_t)m * DD + n];
    }
  }

  // per-row sum / sumsq: 8 cols per lane, shuffle over 16-lane half, LDS over waves
#pragma unroll
  for (int v = 0; v < 8; ++v) {
    float sm = 0.0f, sq = 0.0f;
#pragma unroll
    for (int t = 0; t < 8; ++t) {
      float x = acc[t][v];
      sm += x;
      sq += x * x;
    }
    sm = hsum16(sm);
    sq = hsum16(sq);
    if (col == 0) {
      s1[rbase + v][wave] = sm;
      s2[rbase + v][wave] = sq;
    }
  }
  __syncthreads();

#pragma unroll
  for (int v = 0; v < 8; ++v) {
    int r = rbase + v;
    float sm = s1[r][0] + s1[r][1] + s1[r][2] + s1[r][3];
    float sq = s2[r][0] + s2[r][1] + s2[r][2] + s2[r][3];
    float mu = sm * (1.0f / DD);
    float var = sq * (1.0f / DD) - mu * mu;
    float rstd = rsqrtf(var + 1e-12f);
    int m = m0 + r;
#pragma unroll
    for (int t = 0; t < 8; ++t) {
      int n = n0 + t * 16 + col;
      out[(size_t)m * DD + n] = (acc[t][v] - mu) * rstd * gamma[n] + beta[n];
    }
  }
}

// ------------------------------------------------------------------- launcher
extern "C" void kernel_launch(void* const* d_in, const int* in_sizes, int n_in,
                              void* d_out, int out_size, void* d_ws, size_t ws_size,
                              hipStream_t stream) {
  (void)in_sizes; (void)n_in; (void)out_size; (void)ws_size;

  const float* x_in = (const float*)d_in[0];
  const float* attr = (const float*)d_in[1];   // (B,S,1,D) == flat (B,S,D)
  const float* pos  = (const float*)d_in[2];
  const float* mask = (const float*)d_in[3];
  const float* Wq  = (const float*)d_in[4];  const float* bq  = (const float*)d_in[5];
  const float* Wk  = (const float*)d_in[6];  const float* bk  = (const float*)d_in[7];
  const float* Wv  = (const float*)d_in[8];  const float* bv  = (const float*)d_in[9];
  const float* Wqp = (const float*)d_in[10]; const float* bqp = (const float*)d_in[11];
  const float* Wkp = (const float*)d_in[12]; const float* bkp = (const float*)d_in[13];
  const float* Wqa = (const float*)d_in[14]; const float* bqa = (const float*)d_in[15];
  const float* Wka = (const float*)d_in[16]; const float* bka = (const float*)d_in[17];
  const float* Wd  = (const float*)d_in[18]; const float* bd  = (const float*)d_in[19];
  const float* gamma = (const float*)d_in[20];
  const float* beta  = (const float*)d_in[21];
  const float* wmat  = (const float*)d_in[22];

  char* ws = (char*)d_ws;
  size_t off = 0;
  auto carve = [&](size_t bytes) -> char* {
    char* p = ws + off;
    off += (bytes + 255) & ~(size_t)255;
    return p;
  };
  const size_t MD = (size_t)MM * DD;
  const size_t WSZ = (size_t)DD * DD;
  bf16* Xin  = (bf16*)carve(MD * 2);
  bf16* Xat  = (bf16*)carve(MD * 2);
  bf16* Xpo  = (bf16*)carve(MD * 2);
  bf16* Wt   = (bf16*)carve(8 * WSZ * 2);   // order: q,k,v,qp,kp,qa,ka,d
  bf16* Qi   = (bf16*)carve(MD * 2);
  bf16* Qp   = (bf16*)carve(MD * 2);
  bf16* Qa   = (bf16*)carve(MD * 2);
  bf16* Qcat = (bf16*)carve((size_t)BB * HH * SS * DCAT * 2);
  bf16* Kcat = (bf16*)carve((size_t)BB * HH * SS * DCAT * 2);
  bf16* Vt   = (bf16*)carve(MD * 2);
  bf16* Ctx  = (bf16*)carve(MD * 2);

  int n = (int)MD;
  pack_inputs_kernel<<<(n + 255) / 256, 256, 0, stream>>>(x_in, attr, pos, Xin, Xat, Xpo, n);
  pack_weights_kernel<<<(DD * DD + 255) / 256, 256, 0, stream>>>(Wq, Wk, Wv, Wqp, Wkp, Wqa, Wka, Wd, Wt);

  dim3 gp(MM / 16, DD / 64);
  // Q projections (row-major bf16 outputs, mixed afterwards)
  proj_gemm_kernel<<<gp, 32, 0, stream>>>(Xin, Wt + 0 * WSZ, bq,  Qi, 0, 0);
  proj_gemm_kernel<<<gp, 32, 0, stream>>>(Xpo, Wt + 3 * WSZ, bqp, Qp, 0, 0);
  proj_gemm_kernel<<<gp, 32, 0, stream>>>(Xat, Wt + 5 * WSZ, bqa, Qa, 0, 0);
  // K projections straight into concatenated [B,H,S,192] layout
  proj_gemm_kernel<<<gp, 32, 0, stream>>>(Xin, Wt + 1 * WSZ, bk,  Kcat, 1, 0);
  proj_gemm_kernel<<<gp, 32, 0, stream>>>(Xpo, Wt + 4 * WSZ, bkp, Kcat, 1, 64);
  proj_gemm_kernel<<<gp, 32, 0, stream>>>(Xat, Wt + 6 * WSZ, bka, Kcat, 1, 128);
  // V straight into transposed [B,H,64,S] layout
  proj_gemm_kernel<<<gp, 32, 0, stream>>>(Xin, Wt + 2 * WSZ, bv,  Vt, 2, 0);

  qmix_kernel<<<(n + 255) / 256, 256, 0, stream>>>(Qi, Qp, Qa, wmat, Qcat);

  attention_kernel<<<BB * HH * (SS / 64), 128, 0, stream>>>(Qcat, Kcat, Vt, mask, Ctx);

  out_ln_kernel<<<MM / 16, 128, 0, stream>>>(Ctx, Wt + 7 * WSZ, bd, x_in, gamma, beta,
                                             (float*)d_out);
}
